// PlanePriorNet_55671366091210
// MI455X (gfx1250) — matrix operand order repaired and checked by reading, not compiled, not run on hardware
//
#include <hip/hip_runtime.h>
#include <hip/hip_bf16.h>
#include <math.h>

typedef __attribute__((ext_vector_type(16))) _Float16 v16h;
typedef __attribute__((ext_vector_type(8)))  _Float16 v8h;
typedef __attribute__((ext_vector_type(8)))  float    v8f;

#define P_   256
#define PT_  64
#define N_   (P_ * PT_)   /* 16384 */
#define M_   (2 * N_)     /* 32768 */
#define D_   768
#define KNN_ 9
#define R2_  0.09f

// ---------------- CDNA5 helpers ----------------

__device__ __forceinline__ v8f wmma32(v16h a, v16h b, v8f c) {
  return __builtin_amdgcn_wmma_f32_16x16x32_f16(
      /*neg_a=*/false, a, /*neg_b=*/false, b,
      /*c_mod=*/(short)0, c, /*reuse_a=*/false, /*reuse_b=*/false);
}

// Async global->LDS 16B copy (gfx1250 GLOBAL_LOAD_ASYNC_TO_LDS_B128, ASYNCcnt).
// LDS destination is the low 32 bits of the flat address (LDS aperture offset).
__device__ __forceinline__ void async_copy_b128(const void* gptr, void* lptr) {
  unsigned ldsAddr = (unsigned)(size_t)lptr;
  asm volatile("global_load_async_to_lds_b128 %0, %1, off"
               :: "v"(ldsAddr), "v"(gptr)
               : "memory");
}
__device__ __forceinline__ void wait_async0() {
  asm volatile("s_wait_asynccnt 0x0" ::: "memory");
}

// A fragment: 16x32 f16 tile, row-major with leading dim ld (halves).
// Lane L (0..15): M=L, K=0..7 & 16..23 ; lanes 16..31: M=L-16, K=8..15 & 24..31.
__device__ __forceinline__ v16h ld_afrag(const _Float16* tileBase, int ld) {
  int lane = threadIdx.x & 31;
  const _Float16* p = tileBase + (size_t)(lane & 15) * ld + ((lane >> 4) << 3);
  v8h lo = *(const v8h*)p;
  v8h hi = *(const v8h*)(p + 16);
  return __builtin_shufflevector(lo, hi, 0,1,2,3,4,5,6,7,8,9,10,11,12,13,14,15);
}

// B fragment from transposed weights BT[N][Kpad] (f16).
// Lane L (0..15): col=L, K = kBase+0..7 & 16..23 ; lanes 16..31: col=L-16, K = kBase+8..15 & 24..31.
__device__ __forceinline__ v16h ld_bfrag(const _Float16* BT, int Kpad, int colBase, int kBase) {
  int lane = threadIdx.x & 31;
  const _Float16* p = BT + (size_t)(colBase + (lane & 15)) * Kpad + kBase + ((lane >> 4) << 3);
  v8h lo = *(const v8h*)p;
  v8h hi = *(const v8h*)(p + 16);
  return __builtin_shufflevector(lo, hi, 0,1,2,3,4,5,6,7,8,9,10,11,12,13,14,15);
}

// ---------------- weight prep ----------------

__global__ void cvtA_kernel(const float* __restrict__ src, _Float16* __restrict__ dst, int n) {
  int i = blockIdx.x * blockDim.x + threadIdx.x;
  if (i < n) dst[i] = (_Float16)src[i];
}

// src: f32 [K][Nsrc] row-major; dst: f16 [Ndst][Kpad] (transposed, zero-padded)
__global__ void cvtT_kernel(const float* __restrict__ src, _Float16* __restrict__ dst,
                            int K, int Nsrc, int Kpad, int Ndst) {
  int i = blockIdx.x * blockDim.x + threadIdx.x;
  int total = Kpad * Ndst;
  if (i >= total) return;
  int n = i / Kpad, k = i - n * Kpad;
  float v = (k < K && n < Nsrc) ? src[(size_t)k * Nsrc + n] : 0.0f;
  dst[(size_t)n * Kpad + k] = (_Float16)v;
}

// -------- generic 64x64-tile WMMA GEMM: C(f32) = A(f16[M][K]) * BT(f16[N][K])^T --------
// A is staged per 32-K chunk with double-buffered async global->LDS copies.

__global__ void gemm64_kernel(const _Float16* __restrict__ A, const _Float16* __restrict__ BT,
                              float* __restrict__ C, int Mrows, int Ncols, int K) {
  __shared__ __align__(16) _Float16 ldsA[2][64 * 32];
  int tid = threadIdx.x, wave = tid >> 5, lane = tid & 31;
  int cb = blockIdx.x, rb = blockIdx.y;
  int rowBase = rb * 64, colBase = cb * 64;
  int rt  = wave >> 1;            // row tile 0..3
  int ct0 = (wave & 1) * 2;       // col tiles {ct0, ct0+1}
  // per-thread 16B copy slot: 64 rows x 32 halves = 256 x 8 halves
  int lr = tid >> 2, lk8 = (tid & 3) * 8;
  const _Float16* asrcRow = A + (size_t)(rowBase + lr) * K + lk8;
  _Float16* adst0 = &ldsA[0][lr * 32 + lk8];
  _Float16* adst1 = &ldsA[1][lr * 32 + lk8];
  v8f acc0 = {}, acc1 = {};
  int nChunks = K >> 5;
  async_copy_b128(asrcRow, adst0);
  int buf = 0;
  for (int kc = 0; kc < nChunks; ++kc) {
    wait_async0();
    __syncthreads();
    if (kc + 1 < nChunks)
      async_copy_b128(asrcRow + (size_t)(kc + 1) * 32, buf ? adst0 : adst1);
    v16h a  = ld_afrag(&ldsA[buf][rt * 16 * 32], 32);
    v16h b0 = ld_bfrag(BT, K, colBase + (ct0 + 0) * 16, kc * 32);
    v16h b1 = ld_bfrag(BT, K, colBase + (ct0 + 1) * 16, kc * 32);
    acc0 = wmma32(a, b0, acc0);
    acc1 = wmma32(a, b1, acc1);
    buf ^= 1;
  }
  int col0 = colBase + (ct0 + 0) * 16 + (lane & 15);
  int col1 = colBase + (ct0 + 1) * 16 + (lane & 15);
  int r0 = rowBase + rt * 16 + ((lane >> 4) << 3);
  #pragma unroll
  for (int v = 0; v < 8; ++v) {
    C[(size_t)(r0 + v) * Ncols + col0] = acc0[v];
    C[(size_t)(r0 + v) * Ncols + col1] = acc1[v];
  }
}

// ---------------- stage 1: spatial MLP (771->256->128->3), fused per 64-row group ----------------

__global__ void stage1_kernel(const float* __restrict__ pos, const float* __restrict__ lfs1,
                              const float* __restrict__ sw1, const float* __restrict__ sb1,
                              const _Float16* __restrict__ sw2T, const float* __restrict__ sb2,
                              const _Float16* __restrict__ sw3T, const float* __restrict__ sb3,
                              float* __restrict__ n_pos) {
  __shared__ __align__(16) _Float16 ldsH1[64 * 256];
  __shared__ __align__(16) _Float16 ldsH2[64 * 128];
  __shared__ float ldsPos[64 * 3];
  int tid = threadIdx.x, wave = tid >> 5, lane = tid & 31;
  int g = blockIdx.x;
  int rowBase = g * 64;
  if (tid < 192) ldsPos[tid] = pos[(size_t)rowBase * 3 + tid];
  __syncthreads();
  // h1[r][c] = relu(lf_g.W1head[c] + pos_r.W1tail[c] + b1[c]) ; thread owns column c=tid
  {
    int c = tid;
    float w0 = sw1[768 * 256 + c], w1 = sw1[769 * 256 + c], w2 = sw1[770 * 256 + c];
    float b = sb1[c] + lfs1[(size_t)g * 256 + c];
    for (int r = 0; r < 64; ++r) {
      float v = b + ldsPos[r * 3 + 0] * w0 + ldsPos[r * 3 + 1] * w1 + ldsPos[r * 3 + 2] * w2;
      ldsH1[r * 256 + c] = (_Float16)fmaxf(v, 0.0f);
    }
  }
  __syncthreads();
  // GEMM2: 64 x (K=256) x 128 ; wave -> col tile, 4 row tiles
  {
    v8f acc[4] = {};
    for (int kc = 0; kc < 8; ++kc) {
      v16h b = ld_bfrag(sw2T, 256, wave * 16, kc * 32);
      #pragma unroll
      for (int rt = 0; rt < 4; ++rt) {
        v16h a = ld_afrag(&ldsH1[rt * 16 * 256 + kc * 32], 256);
        acc[rt] = wmma32(a, b, acc[rt]);
      }
    }
    int col = wave * 16 + (lane & 15);
    float bias = sb2[col];
    #pragma unroll
    for (int rt = 0; rt < 4; ++rt) {
      int r0 = rt * 16 + ((lane >> 4) << 3);
      #pragma unroll
      for (int v = 0; v < 8; ++v)
        ldsH2[(r0 + v) * 128 + col] = (_Float16)fmaxf(acc[rt][v] + bias, 0.0f);
    }
  }
  __syncthreads();
  // GEMM3: 64 x (K=128) x 16(pad of 3) ; waves 0..3, then tanh + pos
  if (wave < 4) {
    v8f acc = {};
    for (int kc = 0; kc < 4; ++kc) {
      v16h b = ld_bfrag(sw3T, 128, 0, kc * 32);
      v16h a = ld_afrag(&ldsH2[wave * 16 * 128 + kc * 32], 128);
      acc = wmma32(a, b, acc);
    }
    int col = lane & 15;
    if (col < 3) {
      float bias = sb3[col];
      int r0 = rowBase + wave * 16 + ((lane >> 4) << 3);
      #pragma unroll
      for (int v = 0; v < 8; ++v) {
        float h = fmaxf(acc[v] + bias, 0.0f);
        n_pos[(size_t)(r0 + v) * 3 + col] = tanhf(h) + pos[(size_t)(r0 + v) * 3 + col];
      }
    }
  }
}

// ---------------- stage 2: radius-masked 9-NN per 128-point group ----------------

__global__ void knn_kernel(const float* __restrict__ pos, const float* __restrict__ n_pos,
                           int* __restrict__ idxOut, int* __restrict__ cntOut) {
  __shared__ float sx[128], sy[128], sz[128];
  int p = blockIdx.x, tid = threadIdx.x;
  {
    int j = tid;
    const float* q = (j < 64) ? &pos[(size_t)(p * 64 + j) * 3]
                              : &n_pos[(size_t)(p * 64 + j - 64) * 3];
    sx[j] = q[0]; sy[j] = q[1]; sz[j] = q[2];
  }
  __syncthreads();
  int i = tid;
  float xi = sx[i], yi = sy[i], zi = sz[i];
  float bd[KNN_]; int bi[KNN_];
  #pragma unroll
  for (int u = 0; u < KNN_; ++u) { bd[u] = 1e30f; bi[u] = 0; }
  for (int j = 0; j < 128; ++j) {
    float dx = sx[j] - xi, dy = sy[j] - yi, dz = sz[j] - zi;
    float d2 = dx * dx + dy * dy + dz * dz;
    if (d2 <= R2_ && d2 < bd[KNN_ - 1]) {
      float d = d2; int id = j;
      #pragma unroll
      for (int u = 0; u < KNN_; ++u) {
        if (d < bd[u]) {
          float td = bd[u]; int ti = bi[u];
          bd[u] = d; bi[u] = id; d = td; id = ti;
        }
      }
    }
  }
  int cnt = 0;
  #pragma unroll
  for (int u = 0; u < KNN_; ++u) {
    if (bd[u] < 1e29f) cnt++;
    idxOut[((size_t)p * 128 + i) * KNN_ + u] = bi[u];
  }
  cntOut[p * 128 + i] = cnt;
}

// ---------------- stage 3: edge MLP 6->64->128 + masked max over 9 neighbors ----------------

__global__ void stage3_kernel(const float* __restrict__ pos, const float* __restrict__ n_pos,
                              const int* __restrict__ idxIn, const int* __restrict__ cntIn,
                              const float* __restrict__ cw1, const float* __restrict__ cb1,
                              const _Float16* __restrict__ cw2T, const float* __restrict__ cb2,
                              _Float16* __restrict__ aggh) {
  __shared__ float sx[128], sy[128], sz[128];
  __shared__ int   nidx[144];
  __shared__ int   ncnt[16];
  __shared__ float sfeat[144 * 6];
  __shared__ float scw1[6 * 64];
  __shared__ float scb1[64];
  __shared__ __align__(16) _Float16 ldsM1[144 * 64];
  __shared__ __align__(16) _Float16 ldsM2[144 * 128];
  int b = blockIdx.x;
  int p = b >> 3;
  int tblk = (b & 7) * 16;
  int tid = threadIdx.x;
  if (tid < 128) {
    int j = tid;
    const float* q = (j < 64) ? &pos[(size_t)(p * 64 + j) * 3]
                              : &n_pos[(size_t)(p * 64 + j - 64) * 3];
    sx[j] = q[0]; sy[j] = q[1]; sz[j] = q[2];
  }
  if (tid < 144) nidx[tid] = idxIn[((size_t)p * 128 + tblk) * KNN_ + tid];
  if (tid < 16)  ncnt[tid] = cntIn[p * 128 + tblk + tid];
  for (int e = tid; e < 6 * 64; e += 256) scw1[e] = cw1[e];
  if (tid < 64)  scb1[tid] = cb1[tid];
  __syncthreads();
  if (tid < 144) {
    int t = tid / KNN_;
    int pi = tblk + t;
    int nj = nidx[tid];
    float nx = sx[nj], ny = sy[nj], nz = sz[nj];
    float* f = &sfeat[tid * 6];
    f[0] = nx; f[1] = ny; f[2] = nz;
    f[3] = nx - sx[pi]; f[4] = ny - sy[pi]; f[5] = nz - sz[pi];
  }
  __syncthreads();
  // m1 = relu(feat @ cw1 + cb1) : 144 x 64
  for (int e = tid; e < 144 * 64; e += 256) {
    int row = e >> 6, c = e & 63;
    const float* f = &sfeat[row * 6];
    float v = scb1[c];
    #pragma unroll
    for (int j = 0; j < 6; ++j) v += f[j] * scw1[j * 64 + c];
    ldsM1[e] = (_Float16)fmaxf(v, 0.0f);
  }
  __syncthreads();
  // m2 = relu(m1 @ cw2 + cb2) with validity mask: 144 x (K=64) x 128
  int wave = tid >> 5, lane = tid & 31;
  {
    v8f acc[9] = {};
    #pragma unroll
    for (int kc = 0; kc < 2; ++kc) {
      v16h bfr = ld_bfrag(cw2T, 64, wave * 16, kc * 32);
      #pragma unroll
      for (int rt = 0; rt < 9; ++rt) {
        v16h a = ld_afrag(&ldsM1[rt * 16 * 64 + kc * 32], 64);
        acc[rt] = wmma32(a, bfr, acc[rt]);
      }
    }
    int col = wave * 16 + (lane & 15);
    float bias = cb2[col];
    #pragma unroll
    for (int rt = 0; rt < 9; ++rt) {
      int r0 = rt * 16 + ((lane >> 4) << 3);
      #pragma unroll
      for (int v = 0; v < 8; ++v) {
        int row = r0 + v;
        int t = row / KNN_, k = row - t * KNN_;
        float val = fmaxf(acc[rt][v] + bias, 0.0f);
        if (k >= ncnt[t]) val = -60000.0f;   // masked (min-ish f16)
        ldsM2[row * 128 + col] = (_Float16)val;
      }
    }
  }
  __syncthreads();
  // masked max over k -> agg[p][tblk+t][c]  (stored f16)
  for (int e = tid; e < 16 * 128; e += 256) {
    int t = e >> 7, c = e & 127;
    float m = -3.0e38f;
    #pragma unroll
    for (int k = 0; k < KNN_; ++k)
      m = fmaxf(m, (float)ldsM2[(t * KNN_ + k) * 128 + c]);
    aggh[((size_t)p * 128 + tblk + t) * 128 + c] = (_Float16)m;
  }
}

// ---------------- stage 4: rot MLP (896->512->256->9) + outputs, 32 rows/block ----------------

__global__ void stage4_kernel(const float* __restrict__ pos, const float* __restrict__ n_pos,
                              const _Float16* __restrict__ aggh, const float* __restrict__ lfm1,
                              const float* __restrict__ mb1,
                              const _Float16* __restrict__ mw1tT,
                              const _Float16* __restrict__ mw2T, const float* __restrict__ mb2,
                              const _Float16* __restrict__ mw3T, const float* __restrict__ mb3,
                              float* __restrict__ outPlane, float* __restrict__ outRC) {
  __shared__ __align__(16) _Float16 ldsA[32 * 128];
  __shared__ __align__(16) _Float16 ldsH1[32 * 512];
  __shared__ __align__(16) _Float16 ldsH2[32 * 256];
  __shared__ float ldsG[32 * 9];
  int tid = threadIdx.x, wave = tid >> 5, lane = tid & 31;
  int rowBase = blockIdx.x * 32;
  int half = (rowBase >= N_) ? 1 : 0;
  int rb = half ? (rowBase - N_) : rowBase;
  int p  = rb >> 6;
  int j0 = (rb & 63) + (half ? 64 : 0);
  // async-stage the per-row agg slice (f16, 32x128 = 8KB): 2x b128 per thread
  const _Float16* asrc = &aggh[((size_t)p * 128 + j0) * 128];
  async_copy_b128(asrc + (size_t)tid * 8, &ldsA[tid * 8]);
  async_copy_b128(asrc + 2048 + (size_t)tid * 8, &ldsA[2048 + tid * 8]);
  wait_async0();
  __syncthreads();
  // GEMM1: 32 x (K=128) x 512 ; lf_g @ mw1head precomputed (lfm1) added in epilogue
  {
    v8f acc[4][2] = {};
    for (int kc = 0; kc < 4; ++kc) {
      v16h a0 = ld_afrag(&ldsA[0 * 16 * 128 + kc * 32], 128);
      v16h a1 = ld_afrag(&ldsA[1 * 16 * 128 + kc * 32], 128);
      #pragma unroll
      for (int cg = 0; cg < 4; ++cg) {
        int ct = wave + cg * 8;
        v16h bfr = ld_bfrag(mw1tT, 128, ct * 16, kc * 32);
        acc[cg][0] = wmma32(a0, bfr, acc[cg][0]);
        acc[cg][1] = wmma32(a1, bfr, acc[cg][1]);
      }
    }
    #pragma unroll
    for (int cg = 0; cg < 4; ++cg) {
      int col = (wave + cg * 8) * 16 + (lane & 15);
      float add = lfm1[(size_t)p * 512 + col] + mb1[col];
      #pragma unroll
      for (int rt = 0; rt < 2; ++rt) {
        int r0 = rt * 16 + ((lane >> 4) << 3);
        #pragma unroll
        for (int v = 0; v < 8; ++v)
          ldsH1[(r0 + v) * 512 + col] = (_Float16)fmaxf(acc[cg][rt][v] + add, 0.0f);
      }
    }
  }
  __syncthreads();
  // GEMM2: 32 x (K=512) x 256  (dominant GEMM)
  {
    v8f acc[2][2] = {};
    for (int kc = 0; kc < 16; ++kc) {
      if (kc + 1 < 16)
        __builtin_prefetch(mw2T + (size_t)((wave * 16) + (lane & 15)) * 512 + (kc + 1) * 32, 0, 1);
      v16h a0 = ld_afrag(&ldsH1[0 * 16 * 512 + kc * 32], 512);
      v16h a1 = ld_afrag(&ldsH1[1 * 16 * 512 + kc * 32], 512);
      #pragma unroll
      for (int cg = 0; cg < 2; ++cg) {
        int ct = wave + cg * 8;
        v16h bfr = ld_bfrag(mw2T, 512, ct * 16, kc * 32);
        acc[cg][0] = wmma32(a0, bfr, acc[cg][0]);
        acc[cg][1] = wmma32(a1, bfr, acc[cg][1]);
      }
    }
    #pragma unroll
    for (int cg = 0; cg < 2; ++cg) {
      int col = (wave + cg * 8) * 16 + (lane & 15);
      float bias = mb2[col];
      #pragma unroll
      for (int rt = 0; rt < 2; ++rt) {
        int r0 = rt * 16 + ((lane >> 4) << 3);
        #pragma unroll
        for (int v = 0; v < 8; ++v)
          ldsH2[(r0 + v) * 256 + col] = (_Float16)fmaxf(acc[cg][rt][v] + bias, 0.0f);
      }
    }
  }
  __syncthreads();
  // GEMM3: 32 x (K=256) x 16(pad of 9) ; waves 0..1
  if (wave < 2) {
    v8f acc = {};
    for (int kc = 0; kc < 8; ++kc) {
      v16h a = ld_afrag(&ldsH2[wave * 16 * 256 + kc * 32], 256);
      v16h bfr = ld_bfrag(mw3T, 256, 0, kc * 32);
      acc = wmma32(a, bfr, acc);
    }
    int col = lane & 15;
    if (col < 9) {
      float bias = mb3[col];
      int r0 = wave * 16 + ((lane >> 4) << 3);
      #pragma unroll
      for (int v = 0; v < 8; ++v)
        ldsG[(r0 + v) * 9 + col] = fmaxf(acc[v] + bias, 0.0f);
    }
  }
  __syncthreads();
  // rot_constrain[n] = R^T R  (R = 3x3 from ldsG row)
  if (tid < 32 * 9) {
    int r = tid / 9, e = tid - r * 9;
    int i = e / 3, j = e - i * 3;
    const float* R = &ldsG[r * 9];
    float s = R[0 * 3 + i] * R[0 * 3 + j] + R[1 * 3 + i] * R[1 * 3 + j] + R[2 * 3 + i] * R[2 * 3 + j];
    outRC[(size_t)(rowBase + r) * 9 + e] = s;
  }
  // plane_init: row gi=n*16+t gets R@npts[t] + pos_c[gi % M]
  #pragma unroll
  for (int q = 0; q < 2; ++q) {
    int e = q * 256 + tid;          // 0..511 = 32 rows * 16 points
    int r = e >> 4, t = e & 15;
    int n = rowBase + r;
    const float* R = &ldsG[r * 9];
    float px = -0.2f + (float)(t & 3) * (0.4f / 3.0f);
    float py = -0.2f + (float)(t >> 2) * (0.4f / 3.0f);
    long long gi = (long long)n * 16 + t;
    int ci = (int)(gi % M_);
    const float* cptr = (ci < N_) ? &pos[(size_t)ci * 3] : &n_pos[(size_t)(ci - N_) * 3];
    #pragma unroll
    for (int c = 0; c < 3; ++c) {
      float d = R[c * 3 + 0] * px + R[c * 3 + 1] * py;   // npts z == 0
      outPlane[(size_t)gi * 3 + c] = d + cptr[c];
    }
  }
}

// ---------------- host ----------------

extern "C" void kernel_launch(void* const* d_in, const int* in_sizes, int n_in,
                              void* d_out, int out_size, void* d_ws, size_t ws_size,
                              hipStream_t stream) {
  (void)in_sizes; (void)n_in; (void)out_size; (void)ws_size;
  const float* pos       = (const float*)d_in[1];
  const float* local_fea = (const float*)d_in[4];
  const float* sw1 = (const float*)d_in[5];
  const float* sb1 = (const float*)d_in[6];
  const float* sw2 = (const float*)d_in[7];
  const float* sb2 = (const float*)d_in[8];
  const float* sw3 = (const float*)d_in[9];
  const float* sb3 = (const float*)d_in[10];
  const float* cw1 = (const float*)d_in[11];
  const float* cb1 = (const float*)d_in[12];
  const float* cw2 = (const float*)d_in[13];
  const float* cb2 = (const float*)d_in[14];
  const float* mw1 = (const float*)d_in[15];
  const float* mb1 = (const float*)d_in[16];
  const float* mw2 = (const float*)d_in[17];
  const float* mb2 = (const float*)d_in[18];
  const float* mw3 = (const float*)d_in[19];
  const float* mb3 = (const float*)d_in[20];

  char* ws = (char*)d_ws;
  size_t off = 0;
  auto alloc = [&](size_t bytes) -> char* {
    char* pch = ws + off;
    off = (off + bytes + 255) & ~(size_t)255;
    return pch;
  };
  _Float16* lfh    = (_Float16*)alloc((size_t)P_ * 768 * 2);
  _Float16* sw1hT  = (_Float16*)alloc((size_t)256 * 768 * 2);
  _Float16* mw1hT  = (_Float16*)alloc((size_t)512 * 768 * 2);
  _Float16* mw1tT  = (_Float16*)alloc((size_t)512 * 128 * 2);
  _Float16* sw2T   = (_Float16*)alloc((size_t)128 * 256 * 2);
  _Float16* sw3T   = (_Float16*)alloc((size_t)16  * 128 * 2);
  _Float16* cw2T   = (_Float16*)alloc((size_t)128 * 64  * 2);
  _Float16* mw2T   = (_Float16*)alloc((size_t)256 * 512 * 2);
  _Float16* mw3T   = (_Float16*)alloc((size_t)16  * 256 * 2);
  float*    lfs1   = (float*)alloc((size_t)P_ * 256 * 4);
  float*    lfm1   = (float*)alloc((size_t)P_ * 512 * 4);
  float*    n_pos  = (float*)alloc((size_t)N_ * 3 * 4);
  int*      nnidx  = (int*)alloc((size_t)P_ * 128 * KNN_ * 4);
  int*      nncnt  = (int*)alloc((size_t)P_ * 128 * 4);
  _Float16* aggh   = (_Float16*)alloc((size_t)P_ * 128 * 128 * 2);

  auto grid1 = [](int n) { return dim3((unsigned)((n + 255) / 256)); };

  // weight prep (f16, transposed/padded)
  cvtA_kernel<<<grid1(P_ * 768), 256, 0, stream>>>(local_fea, lfh, P_ * 768);
  cvtT_kernel<<<grid1(768 * 256), 256, 0, stream>>>(sw1, sw1hT, 768, 256, 768, 256);
  cvtT_kernel<<<grid1(768 * 512), 256, 0, stream>>>(mw1, mw1hT, 768, 512, 768, 512);
  cvtT_kernel<<<grid1(128 * 512), 256, 0, stream>>>(mw1 + 768 * 512, mw1tT, 128, 512, 128, 512);
  cvtT_kernel<<<grid1(256 * 128), 256, 0, stream>>>(sw2, sw2T, 256, 128, 256, 128);
  cvtT_kernel<<<grid1(128 * 16),  256, 0, stream>>>(sw3, sw3T, 128, 3, 128, 16);
  cvtT_kernel<<<grid1(64 * 128),  256, 0, stream>>>(cw2, cw2T, 64, 128, 64, 128);
  cvtT_kernel<<<grid1(512 * 256), 256, 0, stream>>>(mw2, mw2T, 512, 256, 512, 256);
  cvtT_kernel<<<grid1(256 * 16),  256, 0, stream>>>(mw3, mw3T, 256, 9, 256, 16);

  // per-group lf projections: lfs1 = lf @ sw1[0:768], lfm1 = lf @ mw1[0:768]
  gemm64_kernel<<<dim3(256 / 64, 256 / 64), 256, 0, stream>>>(lfh, sw1hT, lfs1, 256, 256, 768);
  gemm64_kernel<<<dim3(512 / 64, 256 / 64), 256, 0, stream>>>(lfh, mw1hT, lfm1, 256, 512, 768);

  // stage 1: spatial MLP -> n_pos
  stage1_kernel<<<P_, 256, 0, stream>>>(pos, lfs1, sw1, sb1, sw2T, sb2, sw3T, sb3, n_pos);

  // stage 2: 9-NN within R
  knn_kernel<<<P_, 128, 0, stream>>>(pos, n_pos, nnidx, nncnt);

  // stage 3: edge MLP + masked max -> agg (f16)
  stage3_kernel<<<P_ * 8, 256, 0, stream>>>(pos, n_pos, nnidx, nncnt, cw1, cb1, cw2T, cb2, aggh);

  // stage 4: rot MLP + plane/rot_constrain outputs
  float* outPlane = (float*)d_out;
  float* outRC    = outPlane + (size_t)16 * M_ * 3;
  stage4_kernel<<<M_ / 32, 256, 0, stream>>>(pos, n_pos, aggh, lfm1, mb1,
                                             mw1tT, mw2T, mb2, mw3T, mb3,
                                             outPlane, outRC);
}
